// empty_GNN_67585605370659
// MI455X (gfx1250) — matrix operand order repaired and checked by reading, not compiled, not run on hardware
//
#include <hip/hip_runtime.h>
#include <hip/hip_bf16.h>

typedef __attribute__((ext_vector_type(16))) _Float16 f16x16;
typedef __attribute__((ext_vector_type(8)))  _Float16 f16x8;
typedef __attribute__((ext_vector_type(8)))  float    f32x8;
typedef __attribute__((ext_vector_type(4)))  float    f32x4;

#define EMB   256
#define EMB2  512
#define NLAYER 5
#define BN_EPS 1e-5f

// ---------------------------------------------------------------------------
// h[i] = atom_emb[x_atom[i]] + chir_emb[x_chir[i]]      (float4 per thread)
// ---------------------------------------------------------------------------
__global__ void h_init_kernel(const int* __restrict__ x_atom,
                              const int* __restrict__ x_chir,
                              const float* __restrict__ atom_emb,
                              const float* __restrict__ chir_emb,
                              float* __restrict__ h, int n) {
  int idx  = blockIdx.x * blockDim.x + threadIdx.x;
  int node = idx >> 6;                 // 64 threads per 256-wide row
  int c4   = (idx & 63) << 2;
  if (node >= n) return;
  int a = x_atom[node], c = x_chir[node];
  f32x4 av = *(const f32x4*)(atom_emb + (size_t)a * EMB + c4);
  f32x4 cv = *(const f32x4*)(chir_emb + (size_t)c * EMB + c4);
  *(f32x4*)(h + (size_t)node * EMB + c4) = av + cv;
}

// ---------------------------------------------------------------------------
// f32 -> f16 weight conversion (once per call)
// ---------------------------------------------------------------------------
__global__ void cvt_f16_kernel(const float* __restrict__ src,
                               _Float16* __restrict__ dst, int n) {
  int i = blockIdx.x * blockDim.x + threadIdx.x;
  if (i < n) dst[i] = (_Float16)src[i];
}

// ---------------------------------------------------------------------------
// agg[i] = h[i] + bond_emb[SELF_LOOP=4] + dir_emb[0]    (self-loop message)
// ---------------------------------------------------------------------------
__global__ void agg_init_kernel(const float* __restrict__ h,
                                const float* __restrict__ self_bond, // row 4
                                const float* __restrict__ self_dir,  // row 0
                                float* __restrict__ agg, int n) {
  int idx  = blockIdx.x * blockDim.x + threadIdx.x;
  int node = idx >> 6;
  int c4   = (idx & 63) << 2;
  if (node >= n) return;
  f32x4 hv = *(const f32x4*)(h + (size_t)node * EMB + c4);
  f32x4 bv = *(const f32x4*)(self_bond + c4);
  f32x4 dv = *(const f32x4*)(self_dir + c4);
  *(f32x4*)(agg + (size_t)node * EMB + c4) = hv + bv + dv;
}

// ---------------------------------------------------------------------------
// Edge scatter: agg[dst] += h[src] + bond_emb[bond] + dir_emb[dir]
// 32 threads per edge, 8 channels each, f32 global atomics.
// ---------------------------------------------------------------------------
__global__ void scatter_kernel(const float* __restrict__ h,
                               const int* __restrict__ src,
                               const int* __restrict__ dst,
                               const int* __restrict__ bond,
                               const int* __restrict__ dir,
                               const float* __restrict__ bond_emb_l,
                               const float* __restrict__ dir_emb_l,
                               float* __restrict__ agg, int e) {
  int idx  = blockIdx.x * blockDim.x + threadIdx.x;
  int edge = idx >> 5;
  int lane = idx & 31;
  if (edge >= e) return;
  int s  = src[edge], d = dst[edge];
  int bt = bond[edge], dt = dir[edge];
  int c0 = lane * 8;
  f32x4 h0 = *(const f32x4*)(h + (size_t)s * EMB + c0);
  f32x4 h1 = *(const f32x4*)(h + (size_t)s * EMB + c0 + 4);
  f32x4 b0 = *(const f32x4*)(bond_emb_l + (size_t)bt * EMB + c0);
  f32x4 b1 = *(const f32x4*)(bond_emb_l + (size_t)bt * EMB + c0 + 4);
  f32x4 d0 = *(const f32x4*)(dir_emb_l + (size_t)dt * EMB + c0);
  f32x4 d1 = *(const f32x4*)(dir_emb_l + (size_t)dt * EMB + c0 + 4);
  f32x4 m0 = h0 + b0 + d0;
  f32x4 m1 = h1 + b1 + d1;
  float* ap = agg + (size_t)d * EMB + c0;
  atomicAdd(ap + 0, m0.x); atomicAdd(ap + 1, m0.y);
  atomicAdd(ap + 2, m0.z); atomicAdd(ap + 3, m0.w);
  atomicAdd(ap + 4, m1.x); atomicAdd(ap + 5, m1.y);
  atomicAdd(ap + 6, m1.z); atomicAdd(ap + 7, m1.w);
}

// ---------------------------------------------------------------------------
// GEMM1: hmid[n,512] = relu(agg[n,256] @ W1^T + b1), f16 out.
// W1 stored [512,256] row-major -> B[k][ncol] = W1[ncol][k] is contiguous in k.
// Wave = one 16(M) x 64(N) tile; K loop in steps of 32 via wmma_f32_16x16x32_f16.
// A-frag layout:               halves j   <- K = 16*h + j  (row = lane%16), same k order as B
// B-frag layout:               halves j   <- K = 16*h + j  (col = lane%16)
// ---------------------------------------------------------------------------
__global__ void __launch_bounds__(256) gemm1_kernel(
    const float* __restrict__ A,        // agg [n,256] f32
    const _Float16* __restrict__ W,     // W1 f16 [512,256]
    const float* __restrict__ bias,     // b1 [512]
    _Float16* __restrict__ Out,         // hmid [n,512] f16
    int n) {
  int mtile = blockIdx.x;
  int wave  = threadIdx.x >> 5;
  int lane  = threadIdx.x & 31;
  int m16   = lane & 15;
  int hh    = lane >> 4;
  int ntb   = wave * 64;

  int row = mtile * 16 + m16;
  if (row >= n) row = n - 1;           // clamp (keeps EXEC full for WMMA)
  const float* arow = A + (size_t)row * EMB;

  f32x8 acc[4] = {};
  for (int kt = 0; kt < EMB; kt += 32) {
    f32x4 fa0 = *(const f32x4*)(arow + kt + hh * 16);
    f32x4 fa1 = *(const f32x4*)(arow + kt + hh * 16 + 4);
    f32x4 fa2 = *(const f32x4*)(arow + kt + hh * 16 + 8);
    f32x4 fa3 = *(const f32x4*)(arow + kt + hh * 16 + 12);
    f16x16 a;
#pragma unroll
    for (int j = 0; j < 4; ++j) {
      a[j]      = (_Float16)fa0[j];
      a[4 + j]  = (_Float16)fa1[j];
      a[8 + j]  = (_Float16)fa2[j];
      a[12 + j] = (_Float16)fa3[j];
    }
#pragma unroll
    for (int j = 0; j < 4; ++j) {
      int ncol = ntb + j * 16 + m16;
      f16x16 b = *(const f16x16*)(W + (size_t)ncol * EMB + kt + hh * 16);
      acc[j] = __builtin_amdgcn_wmma_f32_16x16x32_f16(
          false, a, false, b, (short)0, acc[j], false, false);
    }
  }
#pragma unroll
  for (int j = 0; j < 4; ++j) {
    int ncol = ntb + j * 16 + m16;
    float bv = bias[ncol];
#pragma unroll
    for (int r = 0; r < 8; ++r) {
      int orow = mtile * 16 + r + 8 * hh;   // C: elem r -> M = r + 8*(lane/16)
      if (orow < n) {
        float v = acc[j][r] + bv;
        v = v > 0.f ? v : 0.f;              // ReLU
        Out[(size_t)orow * EMB2 + ncol] = (_Float16)v;
      }
    }
  }
}

// ---------------------------------------------------------------------------
// GEMM2: h2[n,256] = hmid[n,512] @ W2^T + b2, f32 out (into agg buffer).
// ---------------------------------------------------------------------------
__global__ void __launch_bounds__(128) gemm2_kernel(
    const _Float16* __restrict__ A,     // hmid [n,512] f16
    const _Float16* __restrict__ W,     // W2 f16 [256,512]
    const float* __restrict__ bias,     // b2 [256]
    float* __restrict__ Out,            // h2 [n,256] f32
    int n) {
  int mtile = blockIdx.x;
  int wave  = threadIdx.x >> 5;
  int lane  = threadIdx.x & 31;
  int m16   = lane & 15;
  int hh    = lane >> 4;
  int ntb   = wave * 64;

  int row = mtile * 16 + m16;
  if (row >= n) row = n - 1;
  const _Float16* arow = A + (size_t)row * EMB2;

  f32x8 acc[4] = {};
  for (int kt = 0; kt < EMB2; kt += 32) {
    f16x8 lo = *(const f16x8*)(arow + kt + hh * 16);
    f16x8 hi = *(const f16x8*)(arow + kt + hh * 16 + 8);
    f16x16 a;
#pragma unroll
    for (int j = 0; j < 8; ++j) { a[j] = lo[j]; a[8 + j] = hi[j]; }
#pragma unroll
    for (int j = 0; j < 4; ++j) {
      int ncol = ntb + j * 16 + m16;
      f16x16 b = *(const f16x16*)(W + (size_t)ncol * EMB2 + kt + hh * 16);
      acc[j] = __builtin_amdgcn_wmma_f32_16x16x32_f16(
          false, a, false, b, (short)0, acc[j], false, false);
    }
  }
#pragma unroll
  for (int j = 0; j < 4; ++j) {
    int ncol = ntb + j * 16 + m16;
    float bv = bias[ncol];
#pragma unroll
    for (int r = 0; r < 8; ++r) {
      int orow = mtile * 16 + r + 8 * hh;
      if (orow < n)
        Out[(size_t)orow * EMB + ncol] = acc[j][r] + bv;
    }
  }
}

// ---------------------------------------------------------------------------
// Batch norm: zero stats, column sum/sumsq reduction, apply.
// ---------------------------------------------------------------------------
__global__ void zero_stats_kernel(float* __restrict__ stats) {
  stats[threadIdx.x] = 0.f;            // launched with 512 threads
}

__global__ void bn_stats_kernel(const float* __restrict__ h2,
                                float* __restrict__ stats, int n, int chunk) {
  int c  = threadIdx.x;                // 256 threads = 256 columns
  int r0 = blockIdx.x * chunk;
  int r1 = r0 + chunk; if (r1 > n) r1 = n;
  float s = 0.f, s2 = 0.f;
  for (int r = r0; r < r1; ++r) {
    float v = h2[(size_t)r * EMB + c];
    s += v; s2 += v * v;
  }
  atomicAdd(&stats[c], s);
  atomicAdd(&stats[EMB + c], s2);
}

__global__ void bn_apply_kernel(const float* __restrict__ h2,
                                const float* __restrict__ stats,
                                const float* __restrict__ w,
                                const float* __restrict__ b,
                                float* __restrict__ h, int n, int do_relu) {
  int idx  = blockIdx.x * blockDim.x + threadIdx.x;
  int node = idx >> 6;
  int c4   = (idx & 63) << 2;
  if (node >= n) return;
  float inv_n = 1.f / (float)n;
  f32x4 v = *(const f32x4*)(h2 + (size_t)node * EMB + c4);
  f32x4 o;
#pragma unroll
  for (int j = 0; j < 4; ++j) {
    int c = c4 + j;
    float mean = stats[c] * inv_n;
    float var  = stats[EMB + c] * inv_n - mean * mean;
    float x = (v[j] - mean) * __frsqrt_rn(var + BN_EPS) * w[c] + b[c];
    if (do_relu) x = x > 0.f ? x : 0.f;
    o[j] = x;
  }
  *(f32x4*)(h + (size_t)node * EMB + c4) = o;
}

// ---------------------------------------------------------------------------
// Host launcher
// ---------------------------------------------------------------------------
extern "C" void kernel_launch(void* const* d_in, const int* in_sizes, int n_in,
                              void* d_out, int out_size, void* d_ws, size_t ws_size,
                              hipStream_t stream) {
  const int*   x_atom    = (const int*)d_in[0];
  const int*   x_chir    = (const int*)d_in[1];
  const int*   ei        = (const int*)d_in[2];
  const int*   ebond     = (const int*)d_in[3];
  const int*   edir      = (const int*)d_in[4];
  const float* atom_emb  = (const float*)d_in[5];
  const float* chir_emb  = (const float*)d_in[6];
  const float* bond_embs = (const float*)d_in[7];
  const float* dir_embs  = (const float*)d_in[8];
  const float* W1s       = (const float*)d_in[9];
  const float* b1s       = (const float*)d_in[10];
  const float* W2s       = (const float*)d_in[11];
  const float* b2s       = (const float*)d_in[12];
  const float* bn_w      = (const float*)d_in[13];
  const float* bn_b      = (const float*)d_in[14];

  const int n = in_sizes[0];
  const int e = in_sizes[2] / 2;
  const int* esrc = ei;
  const int* edst = ei + e;

  float* h = (float*)d_out;

  // Workspace layout (256B-aligned slices)
  auto al = [](size_t x) { return (x + 255) & ~(size_t)255; };
  char* ws = (char*)d_ws;
  size_t off = 0;
  float*    agg  = (float*)(ws + off);    off += al((size_t)n * EMB * sizeof(float));
  _Float16* hmid = (_Float16*)(ws + off); off += al((size_t)n * EMB2 * sizeof(_Float16));
  _Float16* w1f  = (_Float16*)(ws + off); off += al((size_t)NLAYER * EMB2 * EMB * sizeof(_Float16));
  _Float16* w2f  = (_Float16*)(ws + off); off += al((size_t)NLAYER * EMB * EMB2 * sizeof(_Float16));
  float*    stats= (float*)(ws + off);    off += al(2 * EMB * sizeof(float));

  const int wtot = NLAYER * EMB2 * EMB;
  cvt_f16_kernel<<<(wtot + 255) / 256, 256, 0, stream>>>(W1s, w1f, wtot);
  cvt_f16_kernel<<<(wtot + 255) / 256, 256, 0, stream>>>(W2s, w2f, wtot);

  const int elem_blocks = (n * 64 + 255) / 256;    // float4-per-thread grids
  h_init_kernel<<<elem_blocks, 256, 0, stream>>>(x_atom, x_chir, atom_emb, chir_emb, h, n);

  const int mtiles = (n + 15) / 16;
  const int scat_blocks = (e * 32 + 255) / 256;
  const int chunk = (n + 1023) / 1024;

  for (int l = 0; l < NLAYER; ++l) {
    const float* bond_l = bond_embs + (size_t)l * 6 * EMB;
    const float* dir_l  = dir_embs + (size_t)l * 3 * EMB;

    // agg = h + self-loop message (bond row 4, dir row 0)
    agg_init_kernel<<<elem_blocks, 256, 0, stream>>>(h, bond_l + 4 * EMB, dir_l, agg, n);
    // agg[dst] += h[src] + edge embeddings
    scatter_kernel<<<scat_blocks, 256, 0, stream>>>(h, esrc, edst, ebond, edir,
                                                    bond_l, dir_l, agg, e);
    // hmid = relu(agg @ W1^T + b1)
    gemm1_kernel<<<mtiles, 256, 0, stream>>>(agg, w1f + (size_t)l * EMB2 * EMB,
                                             b1s + (size_t)l * EMB2, hmid, n);
    // h2 = hmid @ W2^T + b2   (written into agg buffer)
    gemm2_kernel<<<mtiles, 128, 0, stream>>>(hmid, w2f + (size_t)l * EMB * EMB2,
                                             b2s + (size_t)l * EMB, agg, n);
    // batch norm (+ relu except last layer) -> h
    zero_stats_kernel<<<1, 512, 0, stream>>>(stats);
    bn_stats_kernel<<<1024, 256, 0, stream>>>(agg, stats, n, chunk);
    bn_apply_kernel<<<elem_blocks, 256, 0, stream>>>(agg, stats,
                                                     bn_w + (size_t)l * EMB,
                                                     bn_b + (size_t)l * EMB,
                                                     h, n, (l < NLAYER - 1) ? 1 : 0);
  }
}
